// GCN_Classifier_43284680409778
// MI455X (gfx1250) — matrix operand order repaired and checked by reading, not compiled, not run on hardware
//
#include <hip/hip_runtime.h>
#include <hip/hip_bf16.h>

#define NNODES 100000
#define NEDGES 3200000
#define DIM    512

typedef __attribute__((ext_vector_type(16))) __bf16 v16bf;
typedef __attribute__((ext_vector_type(8)))  __bf16 v8bf;
typedef __attribute__((ext_vector_type(8)))  float  v8f;

// out[M,512] (bf16) = relu?(A[M,512] fp32) @ W[512,512] (fp32) + bias
// Workgroup: 256 thr = 8 waves, tile 128(M) x 64(N); wave tile 32x32 (4 wmma accs).
// Grid: x = N tiles (fast) so the 8 blocks sharing an A-tile run back-to-back -> A
// is read from HBM ~once (L2-resident for siblings); W (1MB) is always L2-resident.
template <int RELU>
__global__ __launch_bounds__(256) void gemm_bias_bf16out_kernel(
    const float* __restrict__ A, const float* __restrict__ W,
    const float* __restrict__ bias, __bf16* __restrict__ out, int M)
{
  // rows padded to 40 bf16 (80B) -> lane bank stride 20 mod 64, conflict-free
  __shared__ __bf16 sA[128][40];
  __shared__ __bf16 sB[64][40];   // B stored transposed: sB[n_local][k_local]

  const int tid   = threadIdx.x;
  const int lane  = tid & 31;
  const int wave  = tid >> 5;
  const int waveM = wave >> 1;    // 0..3
  const int waveN = wave & 1;     // 0..1
  const int n0    = blockIdx.x * 64;
  const int m0    = blockIdx.y * 128;

  v8f acc[2][2] = {};

  for (int k0 = 0; k0 < DIM; k0 += 32) {
    // --- stage A tile: 128x32 fp32 -> bf16 (fused ReLU for layer 2) ---
    // OOB rows are clamped (never stored later) -> unconditional loads, no exec masking.
    #pragma unroll
    for (int j = 0; j < 4; ++j) {
      int q   = tid + 256 * j;       // 1024 float4 total
      int row = q >> 3;              // 8 float4 per 32-wide row
      int c4  = q & 7;
      int gr  = min(m0 + row, M - 1);
      float4 v = *(const float4*)(A + (size_t)gr * DIM + k0 + c4 * 4);
      if (RELU) {
        v.x = fmaxf(v.x, 0.f); v.y = fmaxf(v.y, 0.f);
        v.z = fmaxf(v.z, 0.f); v.w = fmaxf(v.w, 0.f);
      }
      __bf16* p = &sA[row][c4 * 4];
      p[0] = (__bf16)v.x; p[1] = (__bf16)v.y;
      p[2] = (__bf16)v.z; p[3] = (__bf16)v.w;
    }
    // --- stage B tile transposed: W[k0+.., n0+..] -> sB[n][k] ---
    #pragma unroll
    for (int j = 0; j < 2; ++j) {
      int q   = tid + 256 * j;       // 512 float4 total
      int row = q >> 4;              // k local (16 float4 per 64-wide row)
      int c4  = q & 15;
      float4 v = *(const float4*)(W + (size_t)(k0 + row) * DIM + n0 + c4 * 4);
      sB[c4 * 4 + 0][row] = (__bf16)v.x;
      sB[c4 * 4 + 1][row] = (__bf16)v.y;
      sB[c4 * 4 + 2][row] = (__bf16)v.z;
      sB[c4 * 4 + 3][row] = (__bf16)v.w;
    }
    __syncthreads();

    // --- build fragments per ISA 7.12.2 layouts, issue WMMA ---
    const int r  = lane & 15;
    const int kk = (lane >> 4) * 8;    // A: K = kk..kk+7, kk+16..kk+23
    const int kb = (lane >> 4) * 8;    // B: K = kb..kb+7, kb+16..kb+23 (transposed LDS)
    v16bf afrag[2], bfrag[2];
    #pragma unroll
    for (int t = 0; t < 2; ++t) {
      const __bf16* pa = &sA[waveM * 32 + t * 16 + r][kk];
      #pragma unroll
      for (int i = 0; i < 8; ++i) { afrag[t][i] = pa[i]; afrag[t][i + 8] = pa[16 + i]; }
      const __bf16* pb = &sB[waveN * 32 + t * 16 + r][kb];
      #pragma unroll
      for (int i = 0; i < 8; ++i) { bfrag[t][i] = pb[i]; bfrag[t][i + 8] = pb[16 + i]; }
    }
    #pragma unroll
    for (int tm = 0; tm < 2; ++tm)
      #pragma unroll
      for (int tn = 0; tn < 2; ++tn)
        acc[tm][tn] = __builtin_amdgcn_wmma_f32_16x16x32_bf16(
            false, afrag[tm], false, bfrag[tn], (short)0, acc[tm][tn], false, false);
    __syncthreads();
  }

  // --- epilogue: +bias, store bf16 (feeds SpMM gather at half bandwidth) ---
  // Block-uniform split: only the last M-block needs per-row guards.
  const int r  = lane & 15;
  const int hi = lane >> 4;
  if (m0 + 128 <= M) {
    #pragma unroll
    for (int tm = 0; tm < 2; ++tm) {
      #pragma unroll
      for (int tn = 0; tn < 2; ++tn) {
        int col  = n0 + waveN * 32 + tn * 16 + r;
        float bv = bias[col];
        __bf16* op = out + (size_t)(m0 + waveM * 32 + tm * 16 + hi * 8) * DIM + col;
        #pragma unroll
        for (int v = 0; v < 8; ++v)                       // C/D layout (ISA 7.12.2)
          op[(size_t)v * DIM] = (__bf16)(acc[tm][tn][v] + bv);
      }
    }
  } else {
    #pragma unroll
    for (int tm = 0; tm < 2; ++tm) {
      #pragma unroll
      for (int tn = 0; tn < 2; ++tn) {
        int col  = n0 + waveN * 32 + tn * 16 + r;
        float bv = bias[col];
        #pragma unroll
        for (int v = 0; v < 8; ++v) {
          int row = m0 + waveM * 32 + tm * 16 + v + hi * 8;
          if (row < M) out[(size_t)row * DIM + col] = (__bf16)(acc[tm][tn][v] + bv);
        }
      }
    }
  }
}

// out[dst] += val * h[src] ; h is bf16 [N,512], out is fp32 [N,512] (atomic f32 adds)
__global__ __launch_bounds__(256) void spmm_kernel(
    const __bf16* __restrict__ h, const int* __restrict__ src,
    const int* __restrict__ dst, const float* __restrict__ val,
    float* __restrict__ out, int n_edges)
{
  int e = blockIdx.x * 8 + (threadIdx.x >> 5);   // one wave32 per edge
  if (e >= n_edges) return;
  int lane = threadIdx.x & 31;
  int s = src[e];
  int d = dst[e];
  float w = val[e];
  const v8bf* hrow = (const v8bf*)(h + (size_t)s * DIM);   // 64 x 16B chunks per row
  float* od = out + (size_t)d * DIM;
  v8bf x0 = hrow[lane * 2 + 0];
  v8bf x1 = hrow[lane * 2 + 1];
  int base = lane * 16;
  #pragma unroll
  for (int i = 0; i < 8; ++i) atomicAdd(od + base + i,     w * (float)x0[i]);
  #pragma unroll
  for (int i = 0; i < 8; ++i) atomicAdd(od + base + 8 + i, w * (float)x1[i]);
}

__global__ __launch_bounds__(256) void zero_kernel(float4* __restrict__ p, int n4)
{
  int i = blockIdx.x * 256 + threadIdx.x;
  if (i < n4) p[i] = make_float4(0.f, 0.f, 0.f, 0.f);
}

__global__ __launch_bounds__(256) void relu4_kernel(float4* __restrict__ p, int n4)
{
  int i = blockIdx.x * 256 + threadIdx.x;
  if (i < n4) {
    float4 v = p[i];
    v.x = fmaxf(v.x, 0.f); v.y = fmaxf(v.y, 0.f);
    v.z = fmaxf(v.z, 0.f); v.w = fmaxf(v.w, 0.f);
    p[i] = v;
  }
}

extern "C" void kernel_launch(void* const* d_in, const int* in_sizes, int n_in,
                              void* d_out, int out_size, void* d_ws, size_t ws_size,
                              hipStream_t stream) {
  (void)in_sizes; (void)n_in; (void)out_size; (void)ws_size;
  const float* x       = (const float*)d_in[0];
  const int*   adj_src = (const int*)d_in[1];
  const int*   adj_dst = (const int*)d_in[2];
  const float* adj_val = (const float*)d_in[3];
  const float* W1      = (const float*)d_in[4];
  const float* b1      = (const float*)d_in[5];
  const float* W2      = (const float*)d_in[6];
  const float* b2      = (const float*)d_in[7];
  float* out = (float*)d_out;

  // Workspace: H (bf16, 102.4MB) | S (fp32, 204.8MB)
  const size_t nelem = (size_t)NNODES * DIM;          // 51.2M
  __bf16* H = (__bf16*)d_ws;
  float*  S = (float*)((char*)d_ws + nelem * sizeof(__bf16));

  dim3 ggrid(DIM / 64, (NNODES + 127) / 128);         // N fast -> A-tile L2 reuse
  const int n4      = (int)(nelem / 4);               // 12.8M float4
  const int zblocks = (n4 + 255) / 256;
  const int eblocks = (NEDGES + 7) / 8;

  // Layer 1: H = x@W1+b1 (bf16 out) ; S = SpMM(H) fp32 ; relu fused into next GEMM
  gemm_bias_bf16out_kernel<0><<<ggrid, 256, 0, stream>>>(x, W1, b1, H, NNODES);
  zero_kernel<<<zblocks, 256, 0, stream>>>((float4*)S, n4);
  spmm_kernel<<<eblocks, 256, 0, stream>>>(H, adj_src, adj_dst, adj_val, S, NEDGES);

  // Layer 2: H = relu(S)@W2+b2 (bf16 out) ; out = relu(SpMM(H))
  gemm_bias_bf16out_kernel<1><<<ggrid, 256, 0, stream>>>(S, W2, b2, H, NNODES);
  zero_kernel<<<zblocks, 256, 0, stream>>>((float4*)out, n4);
  spmm_kernel<<<eblocks, 256, 0, stream>>>(H, adj_src, adj_dst, adj_val, out, NEDGES);
  relu4_kernel<<<zblocks, 256, 0, stream>>>((float4*)out, n4);
}
